// PyG_GCN_75273596830237
// MI455X (gfx1250) — matrix operand order repaired and checked by reading, not compiled, not run on hardware
//
#include <hip/hip_runtime.h>
#include <hip/hip_bf16.h>
#include <stdint.h>

typedef __attribute__((ext_vector_type(16))) _Float16 v16h;
typedef __attribute__((ext_vector_type(8)))  _Float16 v8h;
typedef __attribute__((ext_vector_type(8)))  float    v8f;

// exact pointee type the async-to-LDS builtin expects (per hipcc diagnostic):
typedef int async_v4i __attribute__((__vector_size__(4 * sizeof(int))));
typedef __attribute__((address_space(1))) async_v4i* g_v4i_p;   // global src
typedef __attribute__((address_space(3))) async_v4i* l_v4i_p;   // LDS dst

#define GCN_N   100000
#define GCN_E   1600000
#define GCN_FIN 128
#define GCN_HID 128
#define GCN_OUT 64

static inline int cdiv(int a, int b) { return (a + b - 1) / b; }

// ---------------- utility kernels ----------------

__global__ void gcn_zero_f32(float* __restrict__ p, int n) {
    int i = blockIdx.x * blockDim.x + threadIdx.x;
    if (i < n) p[i] = 0.0f;
}

__global__ void gcn_f32_to_f16(const float* __restrict__ s, _Float16* __restrict__ d, int n) {
    int i = blockIdx.x * blockDim.x + threadIdx.x;
    if (i < n) d[i] = (_Float16)s[i];
}

// degree histogram over destination nodes (f32 atomic add, L2-resident)
__global__ void gcn_deg(const int* __restrict__ col, float* __restrict__ deg, int e) {
    int i = blockIdx.x * blockDim.x + threadIdx.x;
    if (i < e) atomicAdd(&deg[col[i]], 1.0f);
}

// in-place deg -> dinv = deg>0 ? rsqrt(max(deg,1)) : 0
__global__ void gcn_dinv(float* __restrict__ deg, int n) {
    int i = blockIdx.x * blockDim.x + threadIdx.x;
    if (i < n) {
        float d = deg[i];
        deg[i] = (d > 0.0f) ? rsqrtf(fmaxf(d, 1.0f)) : 0.0f;
    }
}

// ---------------- W prefragmenter ----------------
// Rearranges W[128 x H] (f32, row-major) into f16 WMMA B-fragment order:
//   Wf[(((nt*4)+kc)*32 + lane)*16 + j] = W[(kc*32 + (lane>>4)*8 + (j&7) + (j>>3)*16)*H + nt*16 + (lane&15)]
// so the GEMM can load each lane's 16-half fragment as contiguous 32 bytes.
template <int H>
__global__ void gcn_pack_w(const float* __restrict__ W, _Float16* __restrict__ Wf, int total) {
    int i = blockIdx.x * blockDim.x + threadIdx.x;
    if (i >= total) return;
    const int j    = i & 15;
    const int lane = (i >> 4) & 31;
    const int kc   = (i >> 9) & 3;
    const int nt   = i >> 11;
    const int k = kc * 32 + ((lane >> 4) << 3) + (j & 7) + ((j >> 3) << 4);
    const int n = nt * 16 + (lane & 15);
    Wf[i] = (_Float16)W[(size_t)k * H + n];
}

// ---------------- WMMA GEMM: Y[nrows x H] = X[nrows x 128] @ W[128 x H] ----------------
// Block = 128 threads (4 waves) handles one 16-row tile. X tile (16x128 f16 = 4KB) is
// staged to LDS via gfx1250 async-to-LDS loads; each wave computes 16x16 output tiles.
// Fragment layouts follow CDNA5 ISA 7.12.2 (wave32).
template <int H>
__global__ __launch_bounds__(128) void gcn_gemm_wmma(const _Float16* __restrict__ X,
                                                     const _Float16* __restrict__ Wf,
                                                     float* __restrict__ Y,
                                                     int nrows) {
    __shared__ alignas(16) _Float16 xtile[16 * 128];

    const int t    = threadIdx.x;
    const int lane = t & 31;
    const int wave = t >> 5;
    const int m0   = blockIdx.x * 16;
    if (m0 >= nrows) return;                     // block-uniform

    // ---- stage X tile: 256 chunks of 16B, 2 per thread ----
#if __has_builtin(__builtin_amdgcn_global_load_async_to_lds_b128)
    #pragma unroll
    for (int c = t; c < 256; c += 128) {
        const int row = c >> 4, off = c & 15;
        const _Float16* gp = X + (size_t)(m0 + row) * 128 + off * 8;
        _Float16*       lp = xtile + row * 128 + off * 8;
        __builtin_amdgcn_global_load_async_to_lds_b128(
            (g_v4i_p)(uintptr_t)gp,
            (l_v4i_p)(unsigned int)(uintptr_t)lp, 0, 0);
    }
  #if __has_builtin(__builtin_amdgcn_s_wait_asynccnt)
    __builtin_amdgcn_s_wait_asynccnt(0);
  #else
    asm volatile("s_wait_asynccnt 0" ::: "memory");
  #endif
#else
    #pragma unroll
    for (int c = t; c < 256; c += 128) {
        const int row = c >> 4, off = c & 15;
        *(v8h*)(xtile + row * 128 + off * 8) =
            *(const v8h*)(X + (size_t)(m0 + row) * 128 + off * 8);
    }
#endif
    __syncthreads();

    const int hlane = lane & 15;                 // 0..15
    const int hhalf = lane >> 4;                 // 0 or 1
    const int aoff  = hhalf << 3;                // A: K base 0 or 8 within 32-chunk

    // ---- A fragments for full K=128 (shared across all column tiles) ----
    v16h afrag[4];
    #pragma unroll
    for (int kc = 0; kc < 4; ++kc) {
        const _Float16* ap = xtile + hlane * 128 + kc * 32 + aoff;
        v8h lo = *(const v8h*)ap;                // K = aoff..aoff+7
        v8h hi = *(const v8h*)(ap + 16);         // K = aoff+16..aoff+23
        afrag[kc] = __builtin_shufflevector(lo, hi, 0, 1, 2, 3, 4, 5, 6, 7,
                                                    8, 9, 10, 11, 12, 13, 14, 15);
    }

    // ---- column tiles ----
    for (int nt = wave; nt < H / 16; nt += 4) {
        v8f c = {0.f, 0.f, 0.f, 0.f, 0.f, 0.f, 0.f, 0.f};
        #pragma unroll
        for (int kc = 0; kc < 4; ++kc) {
            const _Float16* bp = Wf + (((size_t)nt * 4 + kc) * 32 + lane) * 16;
            v8h blo = *(const v8h*)bp;           // contiguous 32B per lane
            v8h bhi = *(const v8h*)(bp + 8);
            v16h b = __builtin_shufflevector(blo, bhi, 0, 1, 2, 3, 4, 5, 6, 7,
                                                       8, 9, 10, 11, 12, 13, 14, 15);
            c = __builtin_amdgcn_wmma_f32_16x16x32_f16(
                    /*neg_a=*/false, afrag[kc], /*neg_b=*/false, b,
                    /*c_mod=*/(short)0, c, /*reuse_a=*/false, /*reuse_b=*/false);
        }
        // C/D layout: VGPR r -> M = r (lanes 0-15) or 8+r (lanes 16-31); N = lane%16
        const int ncol  = nt * 16 + hlane;
        const int mbase = m0 + (hhalf << 3);
        #pragma unroll
        for (int r = 0; r < 8; ++r) {
            Y[(size_t)(mbase + r) * H + ncol] = c[r];
        }
    }
}

// ---------------- edge-parallel aggregation: agg[col] += dinv[row]*dinv[col] * xw[row] ----
// One wave per edge; each lane owns H/32 contiguous floats. xw/agg (51MB) are L2-resident.
template <int H>
__global__ __launch_bounds__(256) void gcn_agg(const float* __restrict__ xw,
                                               const int* __restrict__ row,
                                               const int* __restrict__ col,
                                               const float* __restrict__ dinv,
                                               float* __restrict__ agg,
                                               int e) {
    const int lane = threadIdx.x & 31;
    const int wid  = blockIdx.x * (blockDim.x >> 5) + (threadIdx.x >> 5);
    if (wid >= e) return;
    const int r = row[wid];
    const int c = col[wid];
    const float nrm = dinv[r] * dinv[c];
    if (nrm == 0.0f) return;

    constexpr int EPL = H / 32;   // 4 (H=128) or 2 (H=64) floats per lane
    const float* src = xw  + (size_t)r * H + lane * EPL;
    float*       dst = agg + (size_t)c * H + lane * EPL;

    float v[EPL];
    #pragma unroll
    for (int j = 0; j < EPL; ++j) v[j] = src[j];     // coalesced 128b/64b gather
    #pragma unroll
    for (int j = 0; j < EPL; ++j) atomicAdd(&dst[j], nrm * v[j]);
}

// ---------------- bias + ReLU ----------------
template <int H>
__global__ void gcn_bias_relu_f16(const float* __restrict__ agg, const float* __restrict__ b,
                                  _Float16* __restrict__ out, int total) {
    int i = blockIdx.x * blockDim.x + threadIdx.x;
    if (i < total) out[i] = (_Float16)fmaxf(agg[i] + b[i & (H - 1)], 0.0f);
}

template <int H>
__global__ void gcn_bias_relu_f32(const float* __restrict__ agg, const float* __restrict__ b,
                                  float* __restrict__ out, int total) {
    int i = blockIdx.x * blockDim.x + threadIdx.x;
    if (i < total) out[i] = fmaxf(agg[i] + b[i & (H - 1)], 0.0f);
}

// ---------------- host-side launch ----------------
extern "C" void kernel_launch(void* const* d_in, const int* in_sizes, int n_in,
                              void* d_out, int out_size, void* d_ws, size_t ws_size,
                              hipStream_t stream) {
    const float* feat = (const float*)d_in[0];
    const int*   ei   = (const int*)d_in[1];
    const float* W1   = (const float*)d_in[2];
    const float* b1   = (const float*)d_in[3];
    const float* W2   = (const float*)d_in[4];
    const float* b2   = (const float*)d_in[5];
    const float* W3   = (const float*)d_in[6];
    const float* b3   = (const float*)d_in[7];

    const int* rowi = ei;              // edge_index[0]
    const int* coli = ei + GCN_E;      // edge_index[1]

    // workspace carve-up (all 16B aligned)
    char* ws = (char*)d_ws;
    float*    dinv = (float*)ws;                               ws += (size_t)GCN_N * 4;
    _Float16* xh   = (_Float16*)ws;                            ws += (size_t)GCN_N * GCN_HID * 2;
    float*    xw   = (float*)ws;                               ws += (size_t)GCN_N * GCN_HID * 4;
    float*    agg  = (float*)ws;                               ws += (size_t)GCN_N * GCN_HID * 4;
    _Float16* wf   = (_Float16*)ws;                            // 128*128 f16 fragment buffer

    const int T = 256;

    // --- symmetric-norm coefficients ---
    gcn_zero_f32<<<cdiv(GCN_N, T), T, 0, stream>>>(dinv, GCN_N);
    gcn_deg<<<cdiv(GCN_E, T), T, 0, stream>>>(coli, dinv, GCN_E);
    gcn_dinv<<<cdiv(GCN_N, T), T, 0, stream>>>(dinv, GCN_N);

    // --- features -> f16 ---
    gcn_f32_to_f16<<<cdiv(GCN_N * GCN_FIN, T), T, 0, stream>>>(feat, xh, GCN_N * GCN_FIN);

    // ---------------- layer 1: 128 -> 128 ----------------
    gcn_pack_w<GCN_HID><<<cdiv(GCN_FIN * GCN_HID, T), T, 0, stream>>>(W1, wf, GCN_FIN * GCN_HID);
    gcn_gemm_wmma<GCN_HID><<<cdiv(GCN_N, 16), 128, 0, stream>>>(xh, wf, xw, GCN_N);
    gcn_zero_f32<<<cdiv(GCN_N * GCN_HID, T), T, 0, stream>>>(agg, GCN_N * GCN_HID);
    gcn_agg<GCN_HID><<<cdiv(GCN_E, 8), T, 0, stream>>>(xw, rowi, coli, dinv, agg, GCN_E);
    gcn_bias_relu_f16<GCN_HID><<<cdiv(GCN_N * GCN_HID, T), T, 0, stream>>>(agg, b1, xh, GCN_N * GCN_HID);

    // ---------------- layer 2: 128 -> 128 ----------------
    gcn_pack_w<GCN_HID><<<cdiv(GCN_HID * GCN_HID, T), T, 0, stream>>>(W2, wf, GCN_HID * GCN_HID);
    gcn_gemm_wmma<GCN_HID><<<cdiv(GCN_N, 16), 128, 0, stream>>>(xh, wf, xw, GCN_N);
    gcn_zero_f32<<<cdiv(GCN_N * GCN_HID, T), T, 0, stream>>>(agg, GCN_N * GCN_HID);
    gcn_agg<GCN_HID><<<cdiv(GCN_E, 8), T, 0, stream>>>(xw, rowi, coli, dinv, agg, GCN_E);
    gcn_bias_relu_f16<GCN_HID><<<cdiv(GCN_N * GCN_HID, T), T, 0, stream>>>(agg, b2, xh, GCN_N * GCN_HID);

    // ---------------- layer 3: 128 -> 64 ----------------
    gcn_pack_w<GCN_OUT><<<cdiv(GCN_HID * GCN_OUT, T), T, 0, stream>>>(W3, wf, GCN_HID * GCN_OUT);
    gcn_gemm_wmma<GCN_OUT><<<cdiv(GCN_N, 16), 128, 0, stream>>>(xh, wf, xw, GCN_N);
    gcn_zero_f32<<<cdiv(GCN_N * GCN_OUT, T), T, 0, stream>>>(agg, GCN_N * GCN_OUT);
    gcn_agg<GCN_OUT><<<cdiv(GCN_E, 8), T, 0, stream>>>(xw, rowi, coli, dinv, agg, GCN_E);
    gcn_bias_relu_f32<GCN_OUT><<<cdiv(GCN_N * GCN_OUT, T), T, 0, stream>>>(agg, b3, (float*)d_out, GCN_N * GCN_OUT);
}
